// DeprecatedHeteroGAT_7095285973617
// MI455X (gfx1250) — matrix operand order repaired and checked by reading, not compiled, not run on hardware
//
#include <hip/hip_runtime.h>
#include <hip/hip_bf16.h>

typedef __attribute__((ext_vector_type(16))) _Float16 v16h;
typedef __attribute__((ext_vector_type(8)))  float    v8f;

#define NT   200000
#define ND   100000
#define NDEV 4
#define ETD  800000
#define ETT  1600000
#define EDV  800000
#define CH   64

// ---------------- workspace layout (float units) ----------------
// NOTE: O_TD, O_TT, O_DV, HS_TT are deliberately consecutive slabs of NT*CH
// so the WMMA epilogue can compute its destination with pure arithmetic.
static const size_t OFF_O_TD  = 0;
static const size_t OFF_O_TT  = OFF_O_TD  + (size_t)NT * CH;
static const size_t OFF_O_DV  = OFF_O_TT  + (size_t)NT * CH;
static const size_t OFF_HS_TT = OFF_O_DV  + (size_t)NT * CH;
static const size_t OFF_HS_TD = OFF_HS_TT + (size_t)NT * CH;
static const size_t OFF_HS_DV = OFF_HS_TD + (size_t)ND * CH;
static const size_t OFF_AD_TD = OFF_HS_DV + 256;
static const size_t OFF_AD_TT = OFF_AD_TD + NT;
static const size_t OFF_AD_DV = OFF_AD_TT + NT;
static const size_t OFF_AS_TT = OFF_AD_DV + NT;
static const size_t OFF_AS_TD = OFF_AS_TT + NT;
static const size_t OFF_AS_DV = OFF_AS_TD + ND;
static const size_t OFF_M_TD  = OFF_AS_DV + 16;   // uint storage
static const size_t OFF_M_TT  = OFF_M_TD  + NT;
static const size_t OFF_M_DV  = OFF_M_TT  + NT;
static const size_t OFF_DN_TD = OFF_M_DV  + NT;
static const size_t OFF_DN_TT = OFF_DN_TD + NT;
static const size_t OFF_DN_DV = OFF_DN_TT + NT;
static const size_t OFF_EX_TD = OFF_DN_DV + NT;
static const size_t OFF_EX_TT = OFF_EX_TD + ETD;
static const size_t OFF_EX_DV = OFF_EX_TT + ETT;
static const size_t OFF_VECS  = OFF_EX_DV + EDV;   // 9 slots * 16 floats
static const size_t OFF_WT16  = OFF_VECS  + 160;   // 16*32*16 halves = 4096 floats
static const size_t OFF_WD16  = OFF_WT16  + 4096;  // 4*32*16 halves  = 1024 floats
static const size_t OFF_BIAS  = OFF_WD16  + 1024;  // 256 floats (b_td|b_tt|b_dv|0)

// ---------------- helpers ----------------
__device__ __forceinline__ unsigned ordKey(float f) {
    int i = __float_as_int(f);
    return (i < 0) ? ~(unsigned)i : ((unsigned)i | 0x80000000u);
}
__device__ __forceinline__ float ordDecode(unsigned k) {
    int i = (k & 0x80000000u) ? (int)(k & 0x7FFFFFFFu) : ~(int)k;
    return __int_as_float(i);
}
__device__ __forceinline__ void atomAddF(float* p, float v) {
    __hip_atomic_fetch_add(p, v, __ATOMIC_RELAXED, __HIP_MEMORY_SCOPE_AGENT);
}
__device__ __forceinline__ void atomMaxU(unsigned* p, unsigned v) {
    __hip_atomic_fetch_max(p, v, __ATOMIC_RELAXED, __HIP_MEMORY_SCOPE_AGENT);
}

// B-matrix (32x16 f16) per-lane register layout assumed:
//   lane l: n = l&15 ; halves 0..7 -> K = (l>>4)*8 + j ; halves 8..15 -> K = 16 + (l>>4)*8 + (j-8)
// A-matrix (16x32 f16) per ISA table:
//   lane l: m = l&15 ; halves 0..7 -> K = (l>>4)*8 + j ; halves 8..15 -> K = 16 + (l>>4)*8 + (j-8)

// ---------------- K0: small prep (folded vectors, f16 weight packs, bias pack, device rows) ----------------
__global__ void k_prep(const float* Ws_td, const float* as_td, const float* Wd_td, const float* ad_td,
                       const float* We_td, const float* ae_td,
                       const float* Ws_tt, const float* as_tt, const float* Wd_tt, const float* ad_tt,
                       const float* We_tt, const float* ae_tt,
                       const float* Ws_dv, const float* as_dv, const float* Wd_dv, const float* ad_dv,
                       const float* We_dv, const float* ae_dv,
                       const float* Wr_td, const float* Wr_tt, const float* Wr_dv,
                       const float* b_td, const float* b_tt, const float* b_dv,
                       const float* x_dev,
                       float* vecs, _Float16* Wt16, _Float16* Wd16, float* biasAll,
                       float* hs_dv, float* a_src_dv) {
    const float* Wv[9] = {Ws_td, Wd_td, We_td, Ws_tt, Wd_tt, We_tt, Ws_dv, Wd_dv, We_dv};
    const float* Av[9] = {as_td, ad_td, ae_td, as_tt, ad_tt, ae_tt, as_dv, ad_dv, ae_dv};
    const int    Dv[9] = {5, 12, 3, 12, 12, 1, 12, 12, 2};

    // folded attention vectors: vecs[slot*16 + k] = sum_c W[k][c] * a[c]
    for (int t = threadIdx.x; t < 9 * 16; t += blockDim.x) {
        int slot = t >> 4, k = t & 15;
        float r = 0.f;
        if (k < Dv[slot]) {
            for (int c = 0; c < CH; ++c) r += Wv[slot][k * CH + c] * Av[slot][c];
        }
        vecs[t] = r;
    }
    // pack task-side combined weight into per-lane WMMA B order:
    // Wt16[tileN(16)][lane(32)][j(16)], cols: Wr_td|Wr_tt|Wr_dv|Ws_tt, K padded to 32
    for (int t = threadIdx.x; t < 16 * 32 * 16; t += blockDim.x) {
        int j = t & 15, lane = (t >> 4) & 31, tn = t >> 9;
        int k = (j < 8) ? ((lane >> 4) * 8 + j) : (16 + (lane >> 4) * 8 + (j - 8));
        int col = tn * 16 + (lane & 15);
        int g = col >> 6, c = col & 63;
        const float* Wg = (g == 0) ? Wr_td : (g == 1) ? Wr_tt : (g == 2) ? Wr_dv : Ws_tt;
        float v = (k < 12) ? Wg[k * CH + c] : 0.f;
        Wt16[t] = (_Float16)v;
    }
    // pack data-side weight (Ws_td) into Wd16[tileN(4)][lane(32)][j(16)], K padded
    for (int t = threadIdx.x; t < 4 * 32 * 16; t += blockDim.x) {
        int j = t & 15, lane = (t >> 4) & 31, tn = t >> 9;
        int k = (j < 8) ? ((lane >> 4) * 8 + j) : (16 + (lane >> 4) * 8 + (j - 8));
        int col = tn * 16 + (lane & 15);
        float v = (k < 5) ? Ws_td[k * CH + col] : 0.f;
        Wd16[t] = (_Float16)v;
    }
    // combined bias vector over the 256 output columns (last group: hs_tt, no bias)
    for (int t = threadIdx.x; t < 256; t += blockDim.x) {
        int g = t >> 6, c = t & 63;
        biasAll[t] = (g == 0) ? b_td[c] : (g == 1) ? b_tt[c] : (g == 2) ? b_dv[c] : 0.f;
    }
    // device source features hs_dv = x_devices @ Ws_dv  (4 x 64)
    for (int t = threadIdx.x; t < NDEV * CH; t += blockDim.x) {
        int r = t >> 6, c = t & 63;
        float s = 0.f;
        for (int k = 0; k < 12; ++k) s += x_dev[r * 12 + k] * Ws_dv[k * CH + c];
        hs_dv[t] = s;
    }
    __syncthreads();
    if (threadIdx.x < NDEV) {
        float s = 0.f;
        for (int k = 0; k < 12; ++k) s += x_dev[threadIdx.x * 12 + k] * vecs[6 * 16 + k];
        a_src_dv[threadIdx.x] = s;
    }
}

// ---------------- K1: per-task alphas + segment-state init ----------------
__global__ void k_task_alpha(const float* __restrict__ x_tasks, const float* __restrict__ vecs,
                             float* ad_td, float* ad_tt, float* ad_dv, float* as_tt,
                             unsigned* m_td, unsigned* m_tt, unsigned* m_dv,
                             float* dn_td, float* dn_tt, float* dn_dv) {
    int t = blockIdx.x * blockDim.x + threadIdx.x;
    if (t >= NT) return;
    const float* x = x_tasks + (size_t)t * 12;
    float s1 = 0.f, s2 = 0.f, s3 = 0.f, s4 = 0.f;
    for (int k = 0; k < 12; ++k) {
        float xv = x[k];
        s1 += xv * vecs[1 * 16 + k];   // Wd_td @ ad_td
        s2 += xv * vecs[4 * 16 + k];   // Wd_tt @ ad_tt
        s3 += xv * vecs[7 * 16 + k];   // Wd_dv @ ad_dv
        s4 += xv * vecs[3 * 16 + k];   // Ws_tt @ as_tt
    }
    ad_td[t] = s1; ad_tt[t] = s2; ad_dv[t] = s3; as_tt[t] = s4;
    const unsigned NEG_INF_KEY = 0x007FFFFFu;   // ordKey(-inf)
    m_td[t] = NEG_INF_KEY; m_tt[t] = NEG_INF_KEY; m_dv[t] = NEG_INF_KEY;
    dn_td[t] = 0.f; dn_tt[t] = 0.f; dn_dv[t] = 0.f;
}

// ---------------- K2: per-data-node alpha ----------------
__global__ void k_data_alpha(const float* __restrict__ x_data, const float* __restrict__ vecs,
                             float* as_td) {
    int t = blockIdx.x * blockDim.x + threadIdx.x;
    if (t >= ND) return;
    const float* x = x_data + (size_t)t * 5;
    float s = 0.f;
    for (int k = 0; k < 5; ++k) s += x[k] * vecs[0 * 16 + k];
    as_td[t] = s;
}

// ---------------- K3: task transforms via WMMA (LDS-staged A, lane-packed B) ----------------
// output columns 0..255 map to 4 consecutive slabs: res_td|res_tt|res_dv|hs_tt,
// each slab NT*CH floats starting at oBase. Bias folded via biasAll[256].
// 12500 M-tiles of 16 rows (exact), 16 waves/block, wave w = N-tile w.
__global__ void __launch_bounds__(512) k_task_wmma(
        const float* __restrict__ x_tasks, const _Float16* __restrict__ W,
        const float* __restrict__ biasAll, float* __restrict__ oBase) {
    __shared__ __align__(32) _Float16 aLds[32 * 16];
    const int tid   = threadIdx.x;
    const int tileM = blockIdx.x;               // 0..12499

    // Cooperative A staging: entry t -> (lane = t>>4, j = t&15) in WMMA A lane order.
    {
        int lane = tid >> 4, j = tid & 15;
        int m = lane & 15, kh = lane >> 4;
        int k = (j < 8) ? (kh * 8 + j) : (16 + kh * 8 + (j - 8));
        float xv = (k < 12) ? x_tasks[((size_t)tileM * 16 + m) * 12 + k] : 0.f;
        aLds[tid] = (_Float16)xv;
    }
    __syncthreads();

    const int lane = tid & 31;
    const int wv   = tid >> 5;                  // 0..15 -> N tile
    v16h a = *reinterpret_cast<const v16h*>(&aLds[lane * 16]);
    v16h b = *reinterpret_cast<const v16h*>(W + ((size_t)wv * 32 + lane) * 16);

    v8f c = {};
    c = __builtin_amdgcn_wmma_f32_16x16x32_f16(false, a, false, b,
                                               (short)0, c, false, false);

    // Branch-free epilogue: slab + channel from global column index.
    const int gcol = wv * 16 + (lane & 15);     // 0..255
    const float bv = biasAll[gcol];
    float* outp = oBase + (size_t)(gcol >> 6) * ((size_t)NT * CH);
    const int ch = gcol & 63;

    // D layout: VGPR v -> M = v + (lane>=16 ? 8 : 0), N = lane&15
    const int mbase = (lane >> 4) * 8;
    #pragma unroll
    for (int v = 0; v < 8; ++v) {
        int row = tileM * 16 + mbase + v;
        outp[(size_t)row * CH + ch] = c[v] + bv;
    }
}

// ---------------- K4: data transform via WMMA (x_data @ Ws_td) ----------------
__global__ void __launch_bounds__(128) k_data_wmma(
        const float* __restrict__ x_data, const _Float16* __restrict__ W,
        float* __restrict__ hs_td) {
    const int lane  = threadIdx.x & 31;
    const int wv    = threadIdx.x >> 5;         // 0..3 -> N tile
    const int tileM = blockIdx.x;               // 0..6249
    const int m  = lane & 15;
    const int kh = lane >> 4;

    const float* xr = x_data + ((size_t)tileM * 16 + m) * 5;
    v16h a;
    #pragma unroll
    for (int j = 0; j < 8; ++j) {
        int k = kh * 8 + j;
        a[j] = (k < 5) ? (_Float16)xr[k] : (_Float16)0.f;
        a[8 + j] = (_Float16)0.f;               // K = 16..31 padded
    }
    v16h b = *reinterpret_cast<const v16h*>(W + ((size_t)wv * 32 + lane) * 16);

    v8f c = {};
    c = __builtin_amdgcn_wmma_f32_16x16x32_f16(false, a, false, b,
                                               (short)0, c, false, false);

    const int mbase = (lane >> 4) * 8;
    const int ch    = wv * 16 + (lane & 15);
    #pragma unroll
    for (int v = 0; v < 8; ++v) {
        int row = tileM * 16 + mbase + v;
        hs_td[(size_t)row * CH + ch] = c[v];
    }
}

// ---------------- K5: per-edge logits + segment max (ordered-uint atomicMax) ----------------
template <int DE>
__global__ void k_logits(int E, const int* __restrict__ src, const int* __restrict__ dst,
                         const float* __restrict__ eattr,
                         const float* __restrict__ a_src, const float* __restrict__ a_dst,
                         const float* __restrict__ ve,
                         unsigned* __restrict__ mkey, float* __restrict__ lbuf) {
    int e = blockIdx.x * blockDim.x + threadIdx.x;
    if (e >= E) return;
    float l = a_src[src[e]] + a_dst[dst[e]];
    #pragma unroll
    for (int j = 0; j < DE; ++j) l += eattr[(size_t)e * DE + j] * ve[j];
    l = (l > 0.f) ? l : 0.2f * l;               // leaky_relu 0.2
    lbuf[e] = l;
    atomMaxU(&mkey[dst[e]], ordKey(l));
}

// ---------------- K6: ex = exp(l - m[dst]); denom += ex ----------------
__global__ void k_exdenom(int E, const int* __restrict__ dst,
                          const unsigned* __restrict__ mkey,
                          float* __restrict__ lbuf, float* __restrict__ denom) {
    int e = blockIdx.x * blockDim.x + threadIdx.x;
    if (e >= E) return;
    int d = dst[e];
    float ex = __expf(lbuf[e] - ordDecode(mkey[d]));
    lbuf[e] = ex;
    atomAddF(&denom[d], ex);
}

// ---------------- K7: out[dst] += coef * hs[src]  (4 channels/thread) ----------------
__global__ void k_aggregate(int E, const int* __restrict__ src, const int* __restrict__ dst,
                            const float* __restrict__ ex, const float* __restrict__ denom,
                            const float* __restrict__ hs, float* __restrict__ out) {
    int idx = blockIdx.x * blockDim.x + threadIdx.x;
    if (idx >= E * 16) return;
    int e = idx >> 4;
    int q = (idx & 15) * 4;
    int d = dst[e], s = src[e];
    float coef = ex[e] / (denom[d] + 1e-16f);
    const float4 h = *reinterpret_cast<const float4*>(hs + (size_t)s * CH + q);
    float* o = out + (size_t)d * CH + q;
    atomAddF(o + 0, coef * h.x);
    atomAddF(o + 1, coef * h.y);
    atomAddF(o + 2, coef * h.z);
    atomAddF(o + 3, coef * h.w);
}

// ---------------- K8: concat + LayerNorm + leaky(0.01) ----------------
__global__ void __launch_bounds__(64) k_final(
        const float* __restrict__ x_tasks,
        const float* __restrict__ o_td, const float* __restrict__ o_tt, const float* __restrict__ o_dv,
        const float* __restrict__ ln_g, const float* __restrict__ ln_b,
        float* __restrict__ out) {
    const int t = blockIdx.x;
    const int tid = threadIdx.x;
    const int F = 12 + 3 * CH;                  // 204
    float v[4];
    float sum = 0.f, sq = 0.f;
    #pragma unroll
    for (int r = 0; r < 4; ++r) {
        int j = tid + r * 64;
        float x = 0.f;
        if (j < F) {
            if (j < 12)       x = x_tasks[(size_t)t * 12 + j];
            else if (j < 76)  x = o_td[(size_t)t * CH + (j - 12)];
            else if (j < 140) x = o_tt[(size_t)t * CH + (j - 76)];
            else              x = o_dv[(size_t)t * CH + (j - 140)];
            sum += x; sq += x * x;
        }
        v[r] = x;
    }
    // wave32 reduction then cross-wave via LDS
    #pragma unroll
    for (int off = 16; off > 0; off >>= 1) {
        sum += __shfl_xor(sum, off, 32);
        sq  += __shfl_xor(sq,  off, 32);
    }
    __shared__ float s_sum[2], s_sq[2];
    int wv = tid >> 5, lane = tid & 31;
    if (lane == 0) { s_sum[wv] = sum; s_sq[wv] = sq; }
    __syncthreads();
    float tsum = s_sum[0] + s_sum[1];
    float tsq  = s_sq[0]  + s_sq[1];
    float mean = tsum / (float)F;
    float var  = tsq / (float)F - mean * mean;
    float rstd = rsqrtf(var + 1e-5f);
    #pragma unroll
    for (int r = 0; r < 4; ++r) {
        int j = tid + r * 64;
        if (j < F) {
            float y = (v[r] - mean) * rstd * ln_g[j] + ln_b[j];
            y = (y > 0.f) ? y : 0.01f * y;
            out[(size_t)t * F + j] = y;
        }
    }
}

// ---------------- host ----------------
extern "C" void kernel_launch(void* const* d_in, const int* in_sizes, int n_in,
                              void* d_out, int out_size, void* d_ws, size_t ws_size,
                              hipStream_t stream) {
    (void)in_sizes; (void)n_in; (void)out_size; (void)ws_size;
    const float* x_tasks = (const float*)d_in[0];
    const float* x_data  = (const float*)d_in[1];
    const float* x_dev   = (const float*)d_in[2];
    const int*   src_td  = (const int*)d_in[3];
    const int*   dst_td  = (const int*)d_in[4];
    const float* ea_td   = (const float*)d_in[5];
    const int*   src_tt  = (const int*)d_in[6];
    const int*   dst_tt  = (const int*)d_in[7];
    const float* ea_tt   = (const float*)d_in[8];
    const int*   src_dv  = (const int*)d_in[9];
    const int*   dst_dv  = (const int*)d_in[10];
    const float* ea_dv   = (const float*)d_in[11];
    // per-relation params: Ws, Wd, We, as, ad, ae, Wr, b
    const float* Ws_td = (const float*)d_in[12]; const float* Wd_td = (const float*)d_in[13];
    const float* We_td = (const float*)d_in[14]; const float* as_td = (const float*)d_in[15];
    const float* ad_td = (const float*)d_in[16]; const float* ae_td = (const float*)d_in[17];
    const float* Wr_td = (const float*)d_in[18]; const float* b_td  = (const float*)d_in[19];
    const float* Ws_tt = (const float*)d_in[20]; const float* Wd_tt = (const float*)d_in[21];
    const float* We_tt = (const float*)d_in[22]; const float* as_tt = (const float*)d_in[23];
    const float* ad_tt = (const float*)d_in[24]; const float* ae_tt = (const float*)d_in[25];
    const float* Wr_tt = (const float*)d_in[26]; const float* b_tt  = (const float*)d_in[27];
    const float* Ws_dv = (const float*)d_in[28]; const float* Wd_dv = (const float*)d_in[29];
    const float* We_dv = (const float*)d_in[30]; const float* as_dv = (const float*)d_in[31];
    const float* ad_dv = (const float*)d_in[32]; const float* ae_dv = (const float*)d_in[33];
    const float* Wr_dv = (const float*)d_in[34]; const float* b_dv  = (const float*)d_in[35];
    const float* ln_g  = (const float*)d_in[36]; const float* ln_b  = (const float*)d_in[37];

    float* ws = (float*)d_ws;
    float* o_td   = ws + OFF_O_TD;  float* o_tt = ws + OFF_O_TT;  float* o_dv = ws + OFF_O_DV;
    float* hs_tt  = ws + OFF_HS_TT; float* hs_td = ws + OFF_HS_TD; float* hs_dv = ws + OFF_HS_DV;
    float* a_d_td = ws + OFF_AD_TD; float* a_d_tt = ws + OFF_AD_TT; float* a_d_dv = ws + OFF_AD_DV;
    float* a_s_tt = ws + OFF_AS_TT; float* a_s_td = ws + OFF_AS_TD; float* a_s_dv = ws + OFF_AS_DV;
    unsigned* m_td = (unsigned*)(ws + OFF_M_TD);
    unsigned* m_tt = (unsigned*)(ws + OFF_M_TT);
    unsigned* m_dv = (unsigned*)(ws + OFF_M_DV);
    float* dn_td = ws + OFF_DN_TD; float* dn_tt = ws + OFF_DN_TT; float* dn_dv = ws + OFF_DN_DV;
    float* ex_td = ws + OFF_EX_TD; float* ex_tt = ws + OFF_EX_TT; float* ex_dv = ws + OFF_EX_DV;
    float* vecs  = ws + OFF_VECS;
    _Float16* Wt16 = (_Float16*)(ws + OFF_WT16);
    _Float16* Wd16 = (_Float16*)(ws + OFF_WD16);
    float* biasAll = ws + OFF_BIAS;
    float* out = (float*)d_out;

    k_prep<<<1, 256, 0, stream>>>(Ws_td, as_td, Wd_td, ad_td, We_td, ae_td,
                                  Ws_tt, as_tt, Wd_tt, ad_tt, We_tt, ae_tt,
                                  Ws_dv, as_dv, Wd_dv, ad_dv, We_dv, ae_dv,
                                  Wr_td, Wr_tt, Wr_dv, b_td, b_tt, b_dv, x_dev,
                                  vecs, Wt16, Wd16, biasAll, hs_dv, a_s_dv);

    k_task_alpha<<<(NT + 255) / 256, 256, 0, stream>>>(x_tasks, vecs,
                                                       a_d_td, a_d_tt, a_d_dv, a_s_tt,
                                                       m_td, m_tt, m_dv, dn_td, dn_tt, dn_dv);
    k_data_alpha<<<(ND + 255) / 256, 256, 0, stream>>>(x_data, vecs, a_s_td);

    k_task_wmma<<<NT / 16, 512, 0, stream>>>(x_tasks, Wt16, biasAll, o_td);
    k_data_wmma<<<ND / 16, 128, 0, stream>>>(x_data, Wd16, hs_td);

    k_logits<3><<<(ETD + 255) / 256, 256, 0, stream>>>(ETD, src_td, dst_td, ea_td,
                                                       a_s_td, a_d_td, vecs + 2 * 16, m_td, ex_td);
    k_logits<1><<<(ETT + 255) / 256, 256, 0, stream>>>(ETT, src_tt, dst_tt, ea_tt,
                                                       a_s_tt, a_d_tt, vecs + 5 * 16, m_tt, ex_tt);
    k_logits<2><<<(EDV + 255) / 256, 256, 0, stream>>>(EDV, src_dv, dst_dv, ea_dv,
                                                       a_s_dv, a_d_dv, vecs + 8 * 16, m_dv, ex_dv);

    k_exdenom<<<(ETD + 255) / 256, 256, 0, stream>>>(ETD, dst_td, m_td, ex_td, dn_td);
    k_exdenom<<<(ETT + 255) / 256, 256, 0, stream>>>(ETT, dst_tt, m_tt, ex_tt, dn_tt);
    k_exdenom<<<(EDV + 255) / 256, 256, 0, stream>>>(EDV, dst_dv, m_dv, ex_dv, dn_dv);

    k_aggregate<<<(ETD * 16 + 255) / 256, 256, 0, stream>>>(ETD, src_td, dst_td, ex_td, dn_td, hs_td, o_td);
    k_aggregate<<<(ETT * 16 + 255) / 256, 256, 0, stream>>>(ETT, src_tt, dst_tt, ex_tt, dn_tt, hs_tt, o_tt);
    k_aggregate<<<(EDV * 16 + 255) / 256, 256, 0, stream>>>(EDV, src_dv, dst_dv, ex_dv, dn_dv, hs_dv, o_dv);

    k_final<<<NT, 64, 0, stream>>>(x_tasks, o_td, o_tt, o_dv, ln_g, ln_b, out);
}
